// Encoder_50603304681763
// MI455X (gfx1250) — matrix operand order repaired and checked by reading, not compiled, not run on hardware
//
// CDNA5 / gfx1250 bidirectional LSTM encoder.
//   Phase 1: pack emb/W/U to bf16 (W,U transposed to [N][K] for B-fragments).
//   Phase 2: xz = emb[tok] @ W + b via v_wmma_f32_16x16x32_bf16 (store-bound).
//   Phase 3: persistent recurrent kernel: 32 WGs/direction, each owns 16 hidden
//            units. Per step: TDM stages h (and the step's xz slab, overlapped
//            with the GEMM) into LDS; U^T slice staged once by a 3D TDM; h@U
//            via WMMA from LDS; f32 gate math with cell state in VGPRs; device
//            barrier (cluster-barrier fast path) between steps.
#include <hip/hip_runtime.h>
#include <stdint.h>

typedef __attribute__((ext_vector_type(16))) __bf16        bf16x16;
typedef __attribute__((ext_vector_type(8)))  float         f32x8;
typedef __attribute__((ext_vector_type(4)))  unsigned int  u32x4;
typedef __attribute__((ext_vector_type(8)))  int           i32x8;
typedef __attribute__((ext_vector_type(4)))  int           i32x4;

union BFrag {           // 16 bf16 = 32 bytes = two 16B chunks
  bf16x16 v;
  u32x4   q[2];
};

__device__ __forceinline__ uint16_t f32_to_bf16_bits(float f) {
  uint32_t u = __float_as_uint(f);
  u += 0x7FFFu + ((u >> 16) & 1u);          // round-to-nearest-even
  return (uint16_t)(u >> 16);
}
__device__ __forceinline__ float sigmoidf_(float x) {
  return 1.0f / (1.0f + __expf(-x));
}

// ------------------------------------------------------------- TDM descriptors
// D# packing per CDNA5 ISA ch.8: group0 = {flags, lds_addr, global_addr, type},
// group1 = {mask/dsize, tensor dims 0-1, tile dims 0-2, strides}, group2 = dim2.
struct TdmDesc { u32x4 g0; i32x8 g1; i32x4 g2; i32x4 g3; };

__device__ __forceinline__ TdmDesc tdm_make(
    uint32_t lds_addr, uint64_t gaddr, uint32_t dsz_code,
    uint32_t td0, uint32_t td1, uint32_t td2,
    uint32_t tile0, uint32_t tile1, uint32_t tile2,
    uint64_t stride0, uint64_t stride1) {
  TdmDesc t;
  t.g0[0] = 1u;                                    // count=1 valid user D#
  t.g0[1] = lds_addr;                              // LDS byte address
  t.g0[2] = (uint32_t)gaddr;                       // global_addr[31:0]
  t.g0[3] = (uint32_t)((gaddr >> 32) & 0x01FFFFFFull) | (2u << 30); // hi | type=2
  t.g1[0] = (int)(dsz_code << 16);                 // wg_mask=0, data_size
  t.g1[1] = (int)(td0 << 16);                      // tensor_dim0 lo
  t.g1[2] = (int)((td0 >> 16) | (td1 << 16));      // dim0 hi | dim1 lo
  t.g1[3] = (int)((td1 >> 16) | (tile0 << 16));    // dim1 hi | tile_dim0
  t.g1[4] = (int)(tile1 | (tile2 << 16));          // tile_dim1 | tile_dim2
  t.g1[5] = (int)(uint32_t)(stride0 & 0xFFFFFFFFull);
  t.g1[6] = (int)(uint32_t)(((stride0 >> 32) & 0xFFFFull) |
                            ((stride1 & 0xFFFFull) << 16));
  t.g1[7] = (int)(uint32_t)((stride1 >> 16) & 0xFFFFFFFFull);
  t.g2[0] = (int)td2; t.g2[1] = 0; t.g2[2] = 0; t.g2[3] = 0;
  t.g3[0] = 0; t.g3[1] = 0; t.g3[2] = 0; t.g3[3] = 0;
  return t;
}

__device__ __forceinline__ void tdm_issue(const TdmDesc& t) {
#if defined(__clang_major__) && (__clang_major__ >= 23)
  i32x8 z8;
#pragma unroll
  for (int i = 0; i < 8; ++i) z8[i] = 0;
  __builtin_amdgcn_tensor_load_to_lds(t.g0, t.g1, t.g2, t.g3, z8, 0);
#else
  __builtin_amdgcn_tensor_load_to_lds(t.g0, t.g1, t.g2, t.g3, 0);
#endif
}

// ---------------------------------------------------------------- pack kernels
__global__ void k_pack_emb(const float* __restrict__ emb, uint16_t* __restrict__ embb) {
  const size_t i = (size_t)blockIdx.x * 256 + threadIdx.x;
  if (i >= (size_t)34004 * 320) return;
  const int    e = (int)(i % 320);
  const size_t v = i / 320;
  embb[i] = f32_to_bf16_bits(e < 300 ? emb[v * 300 + e] : 0.0f);
}
__global__ void k_pack_w(const float* __restrict__ W_fw, const float* __restrict__ W_bw,
                         uint16_t* __restrict__ Wb) {   // Wb[d][n(2048)][k(320)]
  const size_t i = (size_t)blockIdx.x * 256 + threadIdx.x;
  if (i >= (size_t)2 * 2048 * 320) return;
  const int k = (int)(i % 320);
  const int n = (int)((i / 320) % 2048);
  const int d = (int)(i / ((size_t)320 * 2048));
  const float* W = d ? W_bw : W_fw;
  Wb[i] = f32_to_bf16_bits(k < 300 ? W[(size_t)k * 2048 + n] : 0.0f);
}
__global__ void k_pack_u(const float* __restrict__ U_fw, const float* __restrict__ U_bw,
                         uint16_t* __restrict__ Ub) {   // Ub[d][n(2048)][k(512)]
  const size_t i = (size_t)blockIdx.x * 256 + threadIdx.x;
  if (i >= (size_t)2 * 2048 * 512) return;
  const int k = (int)(i % 512);
  const int n = (int)((i / 512) % 2048);
  const int d = (int)(i / ((size_t)512 * 2048));
  const float* U = d ? U_bw : U_fw;
  Ub[i] = f32_to_bf16_bits(U[(size_t)k * 2048 + n]);
}

// ------------------------------------------------- input projection (both dirs)
__global__ __launch_bounds__(256) void k_proj(
    const int* __restrict__ tokens, const uint16_t* __restrict__ embb,
    const uint16_t* __restrict__ Wb, const float* __restrict__ b_fw,
    const float* __restrict__ b_bw, float* __restrict__ xz) {
  const int lane = threadIdx.x & 31;
  const int wave = blockIdx.x * 8 + (threadIdx.x >> 5);
  const int ng = wave & 31;            // 32 groups of 64 cols
  const int mt = (wave >> 5) & 2047;   // 2048 M tiles of 16 rows
  const int d  = (wave >> 16) & 1;

  const int r0   = mt * 16;
  const int rowA = r0 + (lane & 15);
  const int t = rowA >> 6, b = rowA & 63;
  const int tsrc = d ? (511 - t) : t;
  const int tok  = tokens[b * 512 + tsrc];
  const uint16_t* __restrict__ arow = embb + (size_t)tok * 320;

  const uint16_t* __restrict__ Wd = Wb + (size_t)d * 2048 * 320;
  const int n0 = ng * 64;
  const int kselA = (lane >> 4) << 3;
  const int kselB = (lane >> 4) << 3;

  const uint16_t* __restrict__ Brow[4];
#pragma unroll
  for (int j = 0; j < 4; ++j)
    Brow[j] = Wd + (size_t)(n0 + j * 16 + (lane & 15)) * 320 + kselB;

  f32x8 acc[4];
#pragma unroll
  for (int j = 0; j < 4; ++j) acc[j] = (f32x8){0.f,0.f,0.f,0.f,0.f,0.f,0.f,0.f};

#pragma unroll 2
  for (int k0 = 0; k0 < 320; k0 += 32) {
    BFrag a;
    const uint16_t* ap = arow + k0 + kselA;
    a.q[0] = *(const u32x4*)(ap);
    a.q[1] = *(const u32x4*)(ap + 16);
#pragma unroll
    for (int j = 0; j < 4; ++j) {
      BFrag bb;
      bb.q[0] = *(const u32x4*)(Brow[j] + k0);
      bb.q[1] = *(const u32x4*)(Brow[j] + k0 + 16);
      acc[j] = __builtin_amdgcn_wmma_f32_16x16x32_bf16(
          false, a.v, false, bb.v, (short)0, acc[j], false, false);
    }
  }
  const float* __restrict__ bias = d ? b_bw : b_fw;
  const int rbase = r0 + ((lane >> 4) << 3);
#pragma unroll
  for (int j = 0; j < 4; ++j) {
    const int colg = n0 + j * 16 + (lane & 15);
    const float bv = bias[colg];
#pragma unroll
    for (int r = 0; r < 8; ++r) {
      const int rr = rbase + r;
      const int tt = rr >> 6, bb2 = rr & 63;
      xz[(((size_t)d * 512 + tt) * 64 + bb2) * 2048 + colg] = acc[j][r] + bv;
    }
  }
}

// ----------------------------------------------------- per-step device barrier
__device__ __forceinline__ void dir_barrier(unsigned* cnt, unsigned barno) {
  __syncthreads();
  if (threadIdx.x == 0) {
    __threadfence();                   // publish this WG's h slice
    __hip_atomic_fetch_add(cnt, 1u, __ATOMIC_RELEASE, __HIP_MEMORY_SCOPE_AGENT);
    const unsigned target = 32u * barno;
    while (__hip_atomic_load(cnt, __ATOMIC_ACQUIRE, __HIP_MEMORY_SCOPE_AGENT) < target)
      __builtin_amdgcn_s_sleep(2);
  }
  __syncthreads();
  // Cluster fast path: NOP when not dispatched in clusters (ClusterID==0).
  if (__builtin_amdgcn_cluster_id_x() != 0)
    __builtin_amdgcn_s_cluster_barrier();
}

// ------------------------------------------------ persistent recurrent kernel
// grid = 64 blocks: 0-31 forward, 32-63 backward; block = 512 threads (16 waves)
// dynamic LDS: [0,64K) h | [64K,128K) U^T slice | [128K,144K) gates | [144K,160K) xz
__global__ __launch_bounds__(512) void k_lstm(
    const float* __restrict__ h0_fw, const float* __restrict__ c0_fw,
    const float* __restrict__ h0_bw, const float* __restrict__ c0_bw,
    const uint16_t* __restrict__ Ub, const float* __restrict__ xz,
    uint16_t* __restrict__ hbuf, float* __restrict__ out,
    float* __restrict__ hT, float* __restrict__ cT,
    unsigned* __restrict__ counters) {
  const int d  = blockIdx.x >> 5;
  const int wg = blockIdx.x & 31;
  const int j0 = wg * 16;                       // 16 hidden units per WG
  const int tid = threadIdx.x;
  const int lane = tid & 31, wave = tid >> 5;

  extern __shared__ char smem_dyn[];
  uint16_t* h_lds  = (uint16_t*)(smem_dyn);
  uint16_t* u_lds  = (uint16_t*)(smem_dyn + 65536);
  float*    gates  = (float*)(smem_dyn + 131072);
  float*    xz_lds = (float*)(smem_dyn + 147456);
  const uint32_t lds_base = (uint32_t)(size_t)(void*)smem_dyn;

  const uint16_t* __restrict__ Ud = Ub + (size_t)d * 2048 * 512;
  uint16_t* __restrict__ hping    = hbuf + (size_t)d * 2 * 64 * 512;
  const float* __restrict__ xzd   = xz + (size_t)d * 512 * 64 * 2048;
  const float* __restrict__ h0 = d ? h0_bw : h0_fw;
  const float* __restrict__ c0 = d ? c0_bw : c0_fw;
  unsigned* cnt = counters + d;

  // One-shot TDM: stage U^T gate strips (X=512 K, Y=16 rows, Z=4 strips with
  // inter-strip stride of 512 rows) into LDS for the whole scan.
  if (wave == 0) {
    TdmDesc du = tdm_make(lds_base + 65536,
        (uint64_t)(size_t)(const void*)(Ud + (size_t)j0 * 512), 1u,
        512u, 16u, 4u, 512u, 16u, 4u, 512ull, 262144ull);
    tdm_issue(du);
  }

  // Cell state lives in VGPRs: fixed thread<->(b,j) mapping, 2 elems/thread.
  float creg[2];
#pragma unroll
  for (int it = 0; it < 2; ++it) {
    const int idx = tid + it * 512;
    const int b = idx >> 4, j = idx & 15;
    creg[it] = c0[b * 512 + j0 + j];
    hping[b * 512 + j0 + j] = f32_to_bf16_bits(h0[b * 512 + j0 + j]);
  }
  unsigned bar = 0;
  dir_barrier(cnt, ++bar);
  if (wave == 0) __builtin_amdgcn_s_wait_tensorcnt(0);   // U slice resident
  __syncthreads();

  const int m0 = (wave & 3) * 16;               // batch tile
  const int q  = wave >> 2;                     // gate strip 0..3 (i,f,g,o)
  const int kselA = (lane >> 4) << 3;
  const int kselB = (lane >> 4) << 3;
  const int abase = (m0 + (lane & 15)) * 512 + kselA;
  const int bbase = (q * 16 + (lane & 15)) * 512 + kselB;

  for (int s = 0; s < 512; ++s) {
    if (wave == 0) {
      // h: 2D 64x512 bf16 tile -> LDS (needed before GEMM starts).
      TdmDesc dh = tdm_make(lds_base,
          (uint64_t)(size_t)(const void*)(hping + (size_t)(s & 1) * 32768), 1u,
          512u, 64u, 0u, 512u, 64u, 0u, 512ull, 0ull);
      tdm_issue(dh);
      // xz gate slab: X=16 cols, Y=4 strips @512, Z=64 batches @2048 (f32);
      // DMA overlaps the GEMM, waited on just before the gate math.
      TdmDesc dx = tdm_make(lds_base + 147456,
          (uint64_t)(size_t)(const void*)(xzd + (size_t)s * 131072 + j0), 2u,
          16u, 4u, 64u, 16u, 4u, 64u, 512ull, 2048ull);
      tdm_issue(dx);
      __builtin_amdgcn_s_wait_tensorcnt(1);     // h landed; xz may still fly
    }
    __syncthreads();

    f32x8 acc = (f32x8){0.f,0.f,0.f,0.f,0.f,0.f,0.f,0.f};
#pragma unroll 4
    for (int k0 = 0; k0 < 512; k0 += 32) {      // 16 WMMAs fed from LDS
      BFrag a, bfr;
      a.q[0]   = *(const u32x4*)(h_lds + abase + k0);
      a.q[1]   = *(const u32x4*)(h_lds + abase + k0 + 16);
      bfr.q[0] = *(const u32x4*)(u_lds + bbase + k0);
      bfr.q[1] = *(const u32x4*)(u_lds + bbase + k0 + 16);
      acc = __builtin_amdgcn_wmma_f32_16x16x32_bf16(
          false, a.v, false, bfr.v, (short)0, acc, false, false);
    }
    {
      const int rbase = m0 + ((lane >> 4) << 3);
#pragma unroll
      for (int r = 0; r < 8; ++r)
        gates[(rbase + r) * 64 + q * 16 + (lane & 15)] = acc[r];
    }
    if (wave == 0) __builtin_amdgcn_s_wait_tensorcnt(0);  // xz slab in LDS
    __syncthreads();

#pragma unroll
    for (int it = 0; it < 2; ++it) {
      const int idx = tid + it * 512;
      const int b = idx >> 4, j = idx & 15;
      const float zi = gates[b * 64 +      j] + xz_lds[(b * 4 + 0) * 16 + j];
      const float zf = gates[b * 64 + 16 + j] + xz_lds[(b * 4 + 1) * 16 + j];
      const float zg = gates[b * 64 + 32 + j] + xz_lds[(b * 4 + 2) * 16 + j];
      const float zo = gates[b * 64 + 48 + j] + xz_lds[(b * 4 + 3) * 16 + j];
      const float cn = sigmoidf_(zf) * creg[it] + sigmoidf_(zi) * tanhf(zg);
      const float hn = sigmoidf_(zo) * tanhf(cn);
      creg[it] = cn;
      const int tout = d ? (511 - s) : s;
      out[((size_t)b * 512 + tout) * 1024 + (size_t)d * 512 + j0 + j] = hn;
      hping[(size_t)((s + 1) & 1) * 32768 + b * 512 + j0 + j] =
          f32_to_bf16_bits(hn);
      if (s == 511) {
        hT[b * 1024 + d * 512 + j0 + j] = hn;
        cT[b * 1024 + d * 512 + j0 + j] = cn;
      }
    }
    dir_barrier(cnt, ++bar);                    // release h, protect LDS reuse
  }
}

// ---------------------------------------------------------------------- launch
extern "C" void kernel_launch(void* const* d_in, const int* in_sizes, int n_in,
                              void* d_out, int out_size, void* d_ws, size_t ws_size,
                              hipStream_t stream) {
  (void)in_sizes; (void)n_in; (void)out_size; (void)ws_size;
  const int*   tokens = (const int*)  d_in[0];
  const float* h0_fw  = (const float*)d_in[1];
  const float* c0_fw  = (const float*)d_in[2];
  const float* h0_bw  = (const float*)d_in[3];
  const float* c0_bw  = (const float*)d_in[4];
  const float* emb    = (const float*)d_in[5];
  const float* W_fw   = (const float*)d_in[6];
  const float* U_fw   = (const float*)d_in[7];
  const float* b_fw   = (const float*)d_in[8];
  const float* W_bw   = (const float*)d_in[9];
  const float* U_bw   = (const float*)d_in[10];
  const float* b_bw   = (const float*)d_in[11];

  float* out = (float*)d_out;
  float* hT  = out + (size_t)64 * 512 * 1024;
  float* cT  = hT + (size_t)64 * 1024;

  char* ws = (char*)d_ws;
  size_t off = 0;
  auto carve = [&](size_t bytes) -> char* {
    char* p = ws + off;
    off += (bytes + 255) & ~(size_t)255;
    return p;
  };
  unsigned* counters = (unsigned*)carve(256);
  uint16_t* embb  = (uint16_t*)carve((size_t)34004 * 320 * 2);
  uint16_t* Wb    = (uint16_t*)carve((size_t)2 * 2048 * 320 * 2);
  uint16_t* Ubp   = (uint16_t*)carve((size_t)2 * 2048 * 512 * 2);
  uint16_t* hbuf  = (uint16_t*)carve((size_t)2 * 2 * 64 * 512 * 2);
  float*    xzbuf = (float*)   carve((size_t)2 * 512 * 64 * 2048 * 4);

  hipMemsetAsync(counters, 0, 256, stream);

  { size_t n = (size_t)34004 * 320;
    k_pack_emb<<<dim3((unsigned)((n + 255) / 256)), dim3(256), 0, stream>>>(emb, embb); }
  { size_t n = (size_t)2 * 2048 * 320;
    k_pack_w<<<dim3((unsigned)((n + 255) / 256)), dim3(256), 0, stream>>>(W_fw, W_bw, Wb); }
  { size_t n = (size_t)2 * 2048 * 512;
    k_pack_u<<<dim3((unsigned)((n + 255) / 256)), dim3(256), 0, stream>>>(U_fw, U_bw, Ubp); }

  k_proj<<<dim3(16384), dim3(256), 0, stream>>>(tokens, embb, Wb, b_fw, b_bw, xzbuf);

  // 160 KB dynamic LDS (legal on gfx1250: up to 320 KB per workgroup)
  k_lstm<<<dim3(64), dim3(512), 163840, stream>>>(h0_fw, c0_fw, h0_bw, c0_bw,
                                                  Ubp, xzbuf, hbuf, out, hT, cT,
                                                  counters);
}
